// AttentionalGNN_48704929136767
// MI455X (gfx1250) — matrix operand order repaired and checked by reading, not compiled, not run on hardware
//
#include <hip/hip_runtime.h>
#include <math.h>

#define N_NODES 200
#define N_EDGES 4000

typedef __attribute__((ext_vector_type(16))) _Float16 v16h;
typedef __attribute__((ext_vector_type(8)))  _Float16 v8h;
typedef __attribute__((ext_vector_type(8)))  float    v8f;

// ---------------------------------------------------------------------------
// Fragment layouts for V_WMMA_F32_16X16X32_F16 (CDNA5 ISA 7.12.2, wave32):
// A (16x32): lane r=lane&15 -> row, g=lane>>4; elements 0-7 = K[k0+8g .. +7],
//            elements 8-15 = K[k0+16+8g .. +7]  (two contiguous 16B chunks)
// B (32x16): lane r -> col; elements 0-7 = K[k0+8g .. +7], 8-15 = K[k0+16+8g .. +7]
// C/D:       lane r -> N=r; element v -> M = v + 8g
// All operands are pre-packed, zero-padded f16 with rows%16==0, K%32==0, so
// the inner loop is branch-free: 4x global_load_b128 + 1 wmma.
// ---------------------------------------------------------------------------
__global__ __launch_bounds__(32)
void gemm_wmma(const _Float16* __restrict__ A, int lda,   // (Mpad x Kpad) row-major
               const _Float16* __restrict__ B, int ldb,   // (Npad x Kpad) row-major (B^T layout)
               int K,                                     // multiple of 32
               float* __restrict__ Df, int lddf, int M, int N,  // optional f32 out (bounds-checked)
               _Float16* __restrict__ Dh, int lddh,       // optional padded f16 out
               _Float16* __restrict__ Dt, int lddt,       // optional padded f16 transposed out
               const float* __restrict__ bias, float scale, int relu)
{
  const int lane = threadIdx.x;
  const int g = lane >> 4;
  const int r = lane & 15;
  const int m0 = blockIdx.y * 16;
  const int n0 = blockIdx.x * 16;
  const _Float16* Ap = A + (size_t)(m0 + r) * lda + 8 * g;
  const _Float16* Bp = B + (size_t)(n0 + r) * ldb + 8 * g;

  v8f acc = {};
  for (int k0 = 0; k0 < K; k0 += 32) {
    v8h a0 = *(const v8h*)(Ap + k0);
    v8h a1 = *(const v8h*)(Ap + k0 + 16);
    v8h b0 = *(const v8h*)(Bp + k0);
    v8h b1 = *(const v8h*)(Bp + k0 + 16);
    v16h af, bf;
#pragma unroll
    for (int j = 0; j < 8; ++j) {
      af[j] = a0[j]; af[j + 8] = a1[j];
      bf[j] = b0[j]; bf[j + 8] = b1[j];
    }
    acc = __builtin_amdgcn_wmma_f32_16x16x32_f16(false, af, false, bf,
                                                 (short)0, acc, false, false);
  }

  const int n = n0 + r;
  float badd = bias ? bias[n] : 0.f;
#pragma unroll
  for (int v = 0; v < 8; ++v) {
    int m = m0 + v + 8 * g;
    float x = acc[v] * scale + badd;
    if (relu) x = fmaxf(x, 0.f);
    if (Df && m < M && n < N) Df[(size_t)m * lddf + n] = x;
    if (Dh) Dh[(size_t)m * lddh + n] = (_Float16)x;
    if (Dt) Dt[(size_t)n * lddt + m] = (_Float16)x;
  }
}

// f32 -> padded f16 pack (padding pre-zeroed by memset)
__global__ void pack_f16(const float* __restrict__ in, int ldi,
                         _Float16* __restrict__ out, int ldo, int total, int N)
{
  int idx = blockIdx.x * 256 + threadIdx.x;
  if (idx >= total) return;
  int m = idx / N, n = idx - m * N;
  out[(size_t)m * ldo + n] = (_Float16)in[(size_t)m * ldi + n];
}

// Multiplicity structures from edge_index (int32 [2 x 4000]):
//   a(e)=row[row[e]], b(e)=row[col[e]]
//   cnt[j]  = #{m : b(m)==j}           (softmax key weights)
//   W[c][a] = #{e : col[e]==c,a(e)==a} (scatter-add becomes 200x200 matmul)
__global__ void build_idx(const int* __restrict__ ei,
                          float* __restrict__ W, float* __restrict__ cnt)
{
  int e = blockIdx.x * blockDim.x + threadIdx.x;
  if (e >= N_EDGES) return;
  int r = ei[e];            if (r < 0) r = 0; if (r >= N_EDGES) r = N_EDGES - 1;
  int c = ei[N_EDGES + e];  if (c < 0) c = 0; if (c >= N_NODES) c = N_NODES - 1;
  int a = ei[r];            if (a < 0) a = 0; if (a >= N_NODES) a = N_NODES - 1;
  int b = ei[c];            if (b < 0) b = 0; if (b >= N_NODES) b = N_NODES - 1;
  atomicAdd(&cnt[b], 1.0f);
  atomicAdd(&W[c * N_NODES + a], 1.0f);
}

// Count-weighted softmax over one 200-entry f32 score row; emits padded f16
// probability row (exactly reproduces softmax over the 4000 expanded keys).
__global__ void att_softmax(const float* __restrict__ S, const float* __restrict__ cnt,
                            _Float16* __restrict__ Ph)
{
  __shared__ float red[64];
  const int t = threadIdx.x;                    // 64 threads/row
  const int rid = blockIdx.x;                   // 0..799 = h*200 + i
  const int h = rid / N_NODES, i = rid - h * N_NODES;
  const float* row = S + (size_t)rid * N_NODES;
  _Float16* prow = Ph + ((size_t)h * 208 + i) * 224;

  float m = -1e30f;
  for (int j = t; j < N_NODES; j += 64)
    if (cnt[j] > 0.f) m = fmaxf(m, row[j]);
  red[t] = m; __syncthreads();
  for (int s = 32; s > 0; s >>= 1) { if (t < s) red[t] = fmaxf(red[t], red[t + s]); __syncthreads(); }
  m = red[0]; __syncthreads();

  float wloc[4] = {0.f, 0.f, 0.f, 0.f};
  float acc = 0.f;
#pragma unroll
  for (int q = 0; q < 4; ++q) {
    int j = t + 64 * q;
    if (j < N_NODES) {
      float w = (cnt[j] > 0.f) ? cnt[j] * expf(row[j] - m) : 0.f;
      wloc[q] = w; acc += w;
    }
  }
  red[t] = acc; __syncthreads();
  for (int s = 32; s > 0; s >>= 1) { if (t < s) red[t] += red[t + s]; __syncthreads(); }
  float inv = 1.f / red[0];
#pragma unroll
  for (int q = 0; q < 4; ++q) {
    int j = t + 64 * q;
    if (j < N_NODES) prow[j] = (_Float16)(wloc[q] * inv);
  }
}

// log_softmax over the node axis (200) for each of 128 channels.
__global__ void log_softmax_nodes(const float* __restrict__ X, float* __restrict__ out)
{
  __shared__ float red[64];
  const int c = blockIdx.x, t = threadIdx.x;
  float m = -1e30f;
  for (int i = t; i < N_NODES; i += 64) m = fmaxf(m, X[i * 128 + c]);
  red[t] = m; __syncthreads();
  for (int s = 32; s > 0; s >>= 1) { if (t < s) red[t] = fmaxf(red[t], red[t + s]); __syncthreads(); }
  m = red[0]; __syncthreads();
  float acc = 0.f;
  for (int i = t; i < N_NODES; i += 64) acc += expf(X[i * 128 + c] - m);
  red[t] = acc; __syncthreads();
  for (int s = 32; s > 0; s >>= 1) { if (t < s) red[t] += red[t + s]; __syncthreads(); }
  float lz = logf(red[0]);
  for (int i = t; i < N_NODES; i += 64)
    out[i * 128 + c] = X[i * 128 + c] - m - lz;
}

// ---------------------------------------------------------------------------
static void pack(const float* in, int ldi, _Float16* out, int ldo,
                 int M, int N, hipStream_t s)
{
  int total = M * N;
  pack_f16<<<(total + 255) / 256, 256, 0, s>>>(in, ldi, out, ldo, total, N);
}

struct LayerBufs {
  _Float16 *Wlin, *Wq, *Wk, *Wv, *Wo;   // packed f16 weights
  const float *blin, *bq, *bk, *bv, *bo; // f32 biases
};

static void run_layer(const _Float16* Xp,   // padded f16 input (208 x 128)
                      const LayerBufs& L,
                      _Float16* nlin, _Float16* Qn, _Float16* Kn, _Float16* Vt,
                      float* S, _Float16* Ph, _Float16* Ao, _Float16* Opt,
                      const _Float16* Wh, const float* cnt,
                      float* Hf, _Float16* Hh, int relu, hipStream_t st)
{
  const dim3 blk(32);
  const float s128 = 0.08838834764831845f;  // 1/sqrt(128)

  // node_lin = Xp @ lin_w^T + lin_b -> f16 (208 x 128)
  gemm_wmma<<<dim3(8, 13), blk, 0, st>>>(Xp, 128, L.Wlin, 128, 128,
                                         nullptr, 0, 0, 0, nlin, 128, nullptr, 0,
                                         L.blin, 1.f, 0);
  // Qn/Kn = nlin @ w^T + b -> f16 (208 x 512); Vn stored transposed (512 x 224)
  gemm_wmma<<<dim3(32, 13), blk, 0, st>>>(nlin, 128, L.Wq, 128, 128,
                                          nullptr, 0, 0, 0, Qn, 512, nullptr, 0,
                                          L.bq, 1.f, 0);
  gemm_wmma<<<dim3(32, 13), blk, 0, st>>>(nlin, 128, L.Wk, 128, 128,
                                          nullptr, 0, 0, 0, Kn, 512, nullptr, 0,
                                          L.bk, 1.f, 0);
  gemm_wmma<<<dim3(32, 13), blk, 0, st>>>(nlin, 128, L.Wv, 128, 128,
                                          nullptr, 0, 0, 0, nullptr, 0, Vt, 224,
                                          L.bv, 1.f, 0);
  // per-head scores: S_h = scale * Q_h @ K_h^T -> f32 (200 x 200)
  for (int h = 0; h < 4; ++h)
    gemm_wmma<<<dim3(13, 13), blk, 0, st>>>(Qn + h * 128, 512, Kn + h * 128, 512, 128,
                                            S + h * 40000, 200, 200, 200,
                                            nullptr, 0, nullptr, 0,
                                            nullptr, s128, 0);
  // weighted softmax -> padded f16 P (4 x 208 x 224)
  att_softmax<<<800, 64, 0, st>>>(S, cnt, Ph);
  // per-head Ao_h = P_h @ V_h -> f16 (208 x 512 concat)
  for (int h = 0; h < 4; ++h)
    gemm_wmma<<<dim3(8, 13), blk, 0, st>>>(Ph + h * 208 * 224, 224, Vt + h * 128 * 224, 224, 224,
                                           nullptr, 0, 0, 0, Ao + h * 128, 512, nullptr, 0,
                                           nullptr, 1.f, 0);
  // O = Ao @ o_w^T + o_b, stored transposed f16 (128 x 224)
  gemm_wmma<<<dim3(8, 13), blk, 0, st>>>(Ao, 512, L.Wo, 512, 512,
                                         nullptr, 0, 0, 0, nullptr, 0, Opt, 224,
                                         L.bo, 1.f, 0);
  // scatter-add == W @ O; layer1 -> f16 + ReLU, layer2 -> f32
  gemm_wmma<<<dim3(8, 13), blk, 0, st>>>(Wh, 224, Opt, 224, 224,
                                         Hf, 128, 200, 128, Hh, 128, nullptr, 0,
                                         nullptr, 1.f, relu);
}

extern "C" void kernel_launch(void* const* d_in, const int* in_sizes, int n_in,
                              void* d_out, int out_size, void* d_ws, size_t ws_size,
                              hipStream_t stream)
{
  (void)in_sizes; (void)n_in; (void)out_size; (void)ws_size;
  const float* x  = (const float*)d_in[0];
  const int*   ei = (const int*)d_in[1];
  const float* P1[10]; const float* P2[10];
  for (int i = 0; i < 10; ++i) {
    P1[i] = (const float*)d_in[2 + i];
    P2[i] = (const float*)d_in[12 + i];
  }

  // ---- f32 region ----
  float* wsf = (float*)d_ws;
  float* Wf  = wsf;             // 200*200 = 40000
  float* cnt = Wf + 40000;      // 256 (padded)
  float* S   = cnt + 256;       // 800*200 = 160000
  float* H2  = S + 160000;      // 200*128 = 25600
  const size_t F32_TOTAL = 225856;

  // ---- padded f16 region (zeroed each launch) ----
  _Float16* h0   = (_Float16*)(wsf + F32_TOTAL);
  _Float16* Xp   = h0;            // 208*128  = 26624
  _Float16* Wl1  = Xp   + 26624;  // 128*128  = 16384
  _Float16* Wq1  = Wl1  + 16384;  // 512*128  = 65536
  _Float16* Wk1  = Wq1  + 65536;
  _Float16* Wv1  = Wk1  + 65536;
  _Float16* Wo1  = Wv1  + 65536;  // 128*512  = 65536
  _Float16* Wl2  = Wo1  + 65536;
  _Float16* Wq2  = Wl2  + 16384;
  _Float16* Wk2  = Wq2  + 65536;
  _Float16* Wv2  = Wk2  + 65536;
  _Float16* Wo2  = Wv2  + 65536;
  _Float16* Wh   = Wo2  + 65536;  // 208*224  = 46592
  _Float16* nlin = Wh   + 46592;  // 208*128  = 26624
  _Float16* Qn   = nlin + 26624;  // 208*512  = 106496
  _Float16* Kn   = Qn   + 106496;
  _Float16* Vt   = Kn   + 106496; // 512*224  = 114688
  _Float16* Ph   = Vt   + 114688; // 4*208*224= 186368
  _Float16* Ao   = Ph   + 186368; // 208*512  = 106496
  _Float16* Opt  = Ao   + 106496; // 128*224  = 28672
  _Float16* H1h  = Opt  + 28672;  // 208*128  = 26624
  const size_t F16_TOTAL = 1332736;

  // zero counts + all f16 padding (deterministic rebuild each call)
  hipMemsetAsync(Wf, 0, (40000 + 256) * sizeof(float), stream);
  hipMemsetAsync(h0, 0, F16_TOTAL * sizeof(_Float16), stream);

  build_idx<<<(N_EDGES + 255) / 256, 256, 0, stream>>>(ei, Wf, cnt);

  // pack inputs / weights / count matrix to f16
  pack(x, 128, Xp, 128, 200, 128, stream);
  pack(Wf, 200, Wh, 224, 200, 200, stream);
  pack(P1[0], 128, Wl1, 128, 128, 128, stream);
  pack(P1[2], 128, Wq1, 128, 512, 128, stream);
  pack(P1[4], 128, Wk1, 128, 512, 128, stream);
  pack(P1[6], 128, Wv1, 128, 512, 128, stream);
  pack(P1[8], 512, Wo1, 512, 128, 512, stream);
  pack(P2[0], 128, Wl2, 128, 128, 128, stream);
  pack(P2[2], 128, Wq2, 128, 512, 128, stream);
  pack(P2[4], 128, Wk2, 128, 512, 128, stream);
  pack(P2[6], 128, Wv2, 128, 512, 128, stream);
  pack(P2[8], 512, Wo2, 512, 128, 512, stream);

  LayerBufs L1 = { Wl1, Wq1, Wk1, Wv1, Wo1, P1[1], P1[3], P1[5], P1[7], P1[9] };
  LayerBufs L2 = { Wl2, Wq2, Wk2, Wv2, Wo2, P2[1], P2[3], P2[5], P2[7], P2[9] };

  run_layer(Xp,  L1, nlin, Qn, Kn, Vt, S, Ph, Ao, Opt, Wh, cnt, nullptr, H1h, 1, stream);
  run_layer(H1h, L2, nlin, Qn, Kn, Vt, S, Ph, Ao, Opt, Wh, cnt, H2, nullptr, 0, stream);

  log_softmax_nodes<<<128, 64, 0, stream>>>(H2, (float*)d_out);
}
